// Encoder_28896539967914
// MI455X (gfx1250) — matrix operand order repaired and checked by reading, not compiled, not run on hardware
//
#include <hip/hip_runtime.h>
#include <hip/hip_bf16.h>
#include <stdint.h>

// ---------------------------------------------------------------------------
// NRI encoder for MI455X (gfx1250): all GEMMs via V_WMMA_F32_16X16X32_BF16;
// LSTM weights staged to LDS via GLOBAL_LOAD_ASYNC_TO_LDS_B128 (ASYNCcnt);
// gate nonlinearities via native V_TANH_F32.
// ---------------------------------------------------------------------------

typedef __attribute__((ext_vector_type(16))) __bf16        v16bf;
typedef __attribute__((ext_vector_type(8)))  float         v8f;
typedef __attribute__((ext_vector_type(4)))  unsigned int  v4u;

typedef unsigned short u16;
typedef unsigned char  u8;

#define Bq   8
#define Aq   32
#define Tq   50
#define Hq   128
#define Eq   992           // A*(A-1)
#define BEq  (Bq*Eq)       // 7936

union Frag {
  v16bf v;
  u16   u[16];
  v4u   q[2];
};

__device__ __forceinline__ u16 f2bf(float f) {
  unsigned int x = __float_as_uint(f);
  unsigned int r = x + 0x7FFFu + ((x >> 16) & 1u);   // round-to-nearest-even
  return (u16)(r >> 16);
}
__device__ __forceinline__ float bf2f(u16 h) {
  return __uint_as_float(((unsigned int)h) << 16);
}
__device__ __forceinline__ float eluf(float x) { return x > 0.f ? x : (__expf(x) - 1.f); }

// CDNA5 native transcendental tanh (TRANS32 op).  v_nop covers the documented
// 1-instruction TRANS hazard window the compiler can't see through inline asm.
__device__ __forceinline__ float tanh_fast(float x) {
  float y;
  asm volatile("v_tanh_f32 %0, %1\n\tv_nop" : "=v"(y) : "v"(x));
  return y;
}
// sigmoid(x) = 0.5*tanh(x/2) + 0.5  -> one TRANS + one FMA, no division
__device__ __forceinline__ float sig_fast(float x) {
  return __builtin_fmaf(0.5f, tanh_fast(0.5f * x), 0.5f);
}

// ---------------------------------------------------------------------------
// Weight packing: rearrange W (K x N, strides sk/sn in floats) into the CDNA5
// 16-bit B-matrix 32x16 fragment layout, bf16, zero-padded to Kpad.
// Fragment address = (((nt*Ktiles + kt)*32 + lane)*16 + e)
//   lane<16 : N = nt*16+lane,    K = kt*32 + (e<8 ? e : 16 + e-8)      (e = 0..15)
//   lane>=16: N = nt*16+lane-16, K = kt*32 + 8 + (e<8 ? e : 16 + e-8)
// ---------------------------------------------------------------------------
__global__ void pack_w(const float* __restrict__ W, u16* __restrict__ Wp,
                       int K, int N, long sk, long sn, int Kpad) {
  int idx = blockIdx.x * blockDim.x + threadIdx.x;
  int Ktiles = Kpad >> 5;
  int total = (N >> 4) * Ktiles * 512;
  if (idx >= total) return;
  int e    = idx & 15;
  int lane = (idx >> 4) & 31;
  int rest = idx >> 9;
  int kt   = rest % Ktiles;
  int nt   = rest / Ktiles;
  int n = nt * 16 + (lane & 15);
  int k = kt * 32 + ((e >> 3) << 4) + ((lane >> 4) << 3) + (e & 7);
  float v = (k < K) ? W[(long)k * sk + (long)n * sn] : 0.f;
  Wp[idx] = f2bf(v);
}

// traj (rows x 8 f32) -> X0 (rows x 32 bf16, zero padded)
__global__ void build_x0(const float* __restrict__ traj, u16* __restrict__ X0, long rows) {
  long idx = (long)blockIdx.x * blockDim.x + threadIdx.x;
  if (idx >= rows * 32) return;
  int c = idx & 31; long r = idx >> 5;
  X0[idx] = (c < 8) ? f2bf(traj[r * 8 + c]) : (u16)0;
}

// edge mask em[b,k,t] = mask[b,SEND[k],t] & mask[b,RECV[k],t]
__global__ void make_em(const u8* __restrict__ mask, u8* __restrict__ em) {
  long idx = (long)blockIdx.x * blockDim.x + threadIdx.x;
  long total = (long)Bq * Eq * Tq;
  if (idx >= total) return;
  int t = (int)(idx % Tq); long r = idx / Tq;
  int k = (int)(r % Eq);   int b = (int)(r / Eq);
  int i = k / 31, jj = k % 31, j = jj + (jj >= i ? 1 : 0);
  em[idx] = mask[((long)b * Aq + i) * Tq + t] & mask[((long)b * Aq + j) * Tq + t];
}

// ---------------------------------------------------------------------------
// Fused 2-layer MLP:  Y = elu(elu(X@W1+b1)@W2+b2).  K is a compile-time
// constant so the 16xK A-tile is loaded once into registers and reused for
// all 8 N-tiles.  Layer-1 staged through LDS for re-fragmentation; layer-2
// result staged through LDS and flushed as 8 coalesced b128 stores.
// ---------------------------------------------------------------------------
template<int K>
__global__ __launch_bounds__(128)
void mlp2_wmma(const u16* __restrict__ X, long rows,
               const u16* __restrict__ W1p, const float* __restrict__ b1,
               const u16* __restrict__ W2p, const float* __restrict__ b2,
               u16* __restrict__ Y) {
  constexpr int K1t = K / 32;
  __shared__ __align__(16) u16 hbuf[4][16 * Hq];
  const int lane = threadIdx.x & 31;
  const int wave = threadIdx.x >> 5;
  const int half = lane >> 4;
  const int lrow = lane & 15;

  long tiles = rows >> 4;
  long tile = (long)blockIdx.x * 4 + wave;
  if (tile >= tiles) tile = tiles - 1;        // duplicate last tile: benign, keeps EXEC full
  const long row0 = tile << 4;

  const u16* xrow = X + (row0 + lrow) * (long)K;
  u16* hb = hbuf[wave];

  // ---- layer 1: preload all A fragments ----
  Frag af[K1t];
  #pragma unroll
  for (int kt = 0; kt < K1t; ++kt) {
    const v4u* ap = (const v4u*)(xrow + kt * 32 + half * 8);
    af[kt].q[0] = ap[0]; af[kt].q[1] = ap[2];
  }
  #pragma unroll
  for (int nt = 0; nt < 8; ++nt) {
    v8f acc;
    float bias = b1[nt * 16 + lrow];
    #pragma unroll
    for (int i = 0; i < 8; ++i) acc[i] = bias;
    #pragma unroll
    for (int kt = 0; kt < K1t; ++kt) {
      Frag b;
      const v4u* bp = (const v4u*)(W1p + (((size_t)(nt * K1t + kt) * 32 + lane) << 4));
      b.q[0] = bp[0]; b.q[1] = bp[1];
      acc = __builtin_amdgcn_wmma_f32_16x16x32_bf16(false, af[kt].v, false, b.v,
                                                    (short)0, acc, false, false);
    }
    #pragma unroll
    for (int v = 0; v < 8; ++v)
      hb[(v + half * 8) * Hq + nt * 16 + lrow] = f2bf(eluf(acc[v]));
  }

  // ---- layer 2 (K = 128, A from LDS, preloaded once) ----
  const u16* hrow = hb + lrow * Hq;
  Frag hf[4];
  #pragma unroll
  for (int kt = 0; kt < 4; ++kt) {
    const v4u* ap = (const v4u*)(hrow + kt * 32 + half * 8);
    hf[kt].q[0] = ap[0]; hf[kt].q[1] = ap[2];
  }
  #pragma unroll
  for (int nt = 0; nt < 8; ++nt) {
    v8f acc;
    float bias = b2[nt * 16 + lrow];
    #pragma unroll
    for (int i = 0; i < 8; ++i) acc[i] = bias;
    #pragma unroll
    for (int kt = 0; kt < 4; ++kt) {
      Frag b;
      const v4u* bp = (const v4u*)(W2p + (((size_t)(nt * 4 + kt) * 32 + lane) << 4));
      b.q[0] = bp[0]; b.q[1] = bp[1];
      acc = __builtin_amdgcn_wmma_f32_16x16x32_bf16(false, hf[kt].v, false, b.v,
                                                    (short)0, acc, false, false);
    }
    #pragma unroll
    for (int v = 0; v < 8; ++v)
      hb[(v + half * 8) * Hq + nt * 16 + lrow] = f2bf(eluf(acc[v]));
  }
  // coalesced flush: 16x128 bf16 tile is contiguous at Y + row0*128
  {
    const v4u* hq = (const v4u*)hb;          // 256 16-byte chunks
    v4u* yq = (v4u*)(Y + row0 * (long)Hq);
    #pragma unroll
    for (int it = 0; it < 8; ++it)
      yq[it * 32 + lane] = hq[it * 32 + lane];
  }
}

// Xe1[b,k,t, 0:128]=n1[b,SEND[k],t,:], [128:256]=n1[b,RECV[k],t,:]
__global__ void concat_edge1(const u16* __restrict__ n1, u16* __restrict__ Xe1) {
  long idx = (long)blockIdx.x * blockDim.x + threadIdx.x;
  long total = (long)Bq * Eq * Tq * 256;
  if (idx >= total) return;
  int c = (int)(idx & 255); long r = idx >> 8;
  int t = (int)(r % Tq); long rr = r / Tq;
  int k = (int)(rr % Eq); int b = (int)(rr / Eq);
  int i = k / 31, jj = k % 31, j = jj + (jj >= i ? 1 : 0);
  int node = (c < 128) ? i : j;
  Xe1[idx] = n1[(((long)b * Aq + node) * Tq + t) * Hq + (c & 127)];
}

// agg[b,j,t,c] = sum_{i!=j} em ? e1[b,eidx(i,j),t,c] : 0, / 31   (bf16 out)
__global__ void agg_edges(const u16* __restrict__ e1, const u8* __restrict__ em,
                          u16* __restrict__ out) {
  long idx = (long)blockIdx.x * blockDim.x + threadIdx.x;
  long total = (long)Bq * Aq * Tq * Hq;
  if (idx >= total) return;
  int c = (int)(idx & 127); long r = idx >> 7;
  int t = (int)(r % Tq); long rr = r / Tq;
  int j = (int)(rr % Aq); int b = (int)(rr / Aq);
  float s = 0.f;
  for (int i = 0; i < Aq; ++i) {
    if (i == j) continue;
    int k = i * 31 + (j - (j > i ? 1 : 0));
    long er = ((long)b * Eq + k) * Tq + t;
    if (em[er]) s += bf2f(e1[er * Hq + c]);
  }
  out[idx] = f2bf(s * (1.f / 31.f));
}

// Xe2 = concat(n2[send], n2[recv], e1) -> rows x 384
__global__ void concat_edge2(const u16* __restrict__ n2, const u16* __restrict__ e1,
                             u16* __restrict__ Xe2) {
  long idx = (long)blockIdx.x * blockDim.x + threadIdx.x;
  long total = (long)Bq * Eq * Tq * 384;
  if (idx >= total) return;
  int c = (int)(idx % 384); long r = idx / 384;
  int t = (int)(r % Tq); long rr = r / Tq;
  int k = (int)(rr % Eq); int b = (int)(rr / Eq);
  u16 v;
  if (c < 256) {
    int i = k / 31, jj = k % 31, j = jj + (jj >= i ? 1 : 0);
    int node = (c < 128) ? i : j;
    v = n2[(((long)b * Aq + node) * Tq + t) * Hq + (c & 127)];
  } else {
    v = e1[r * Hq + (c - 256)];
  }
  Xe2[idx] = v;
}

// ---------------------------------------------------------------------------
// LSTM: one block per 16 rows of BE; h (bf16) + c (f32) persistent in LDS.
// W_ih^T / W_hh^T (256 KB packed bf16) staged into dynamic LDS ONCE per block
// with GLOBAL_LOAD_ASYNC_TO_LDS_B128 (ASYNCcnt).  Per gate N-tile all 8
// B-fragments are preloaded so the ds_loads pipeline ahead of the WMMA chain.
// Gate order (jnp.split): 0=i, 1=f, 2=g(tanh), 3=o.
// ---------------------------------------------------------------------------
__global__ __launch_bounds__(128)
void lstm_wmma(const u16* __restrict__ Xg,      // (BE,T,128) bf16
               const u8*  __restrict__ em,      // (BE,T)
               const u16* __restrict__ Wihp, const u16* __restrict__ Whhp,
               const float* __restrict__ bih, const float* __restrict__ bhh,
               u16* __restrict__ fe) {          // (BE,T,128) bf16 (h_new, unmasked)
  extern __shared__ u16 wlds[];                 // 131072 u16: [0)=W_ih^T, [65536)=W_hh^T
  __shared__ __align__(16) u16 hst[16 * Hq];
  __shared__ float cst[16 * Hq];
  __shared__ float gbuf[4][16 * Hq];
  __shared__ u8 ems[16];
  const int lane = threadIdx.x & 31;
  const int wave = threadIdx.x >> 5;
  const int half = lane >> 4;
  const int lrow = lane & 15;
  const long row0 = (long)blockIdx.x << 4;

  // ---- async DMA: global -> LDS, 2 x 128 KB of packed weights ----
  {
    unsigned lbase = (unsigned)(size_t)(void*)wlds;
    unsigned long long g0 = (unsigned long long)(size_t)Wihp;
    unsigned long long g1 = (unsigned long long)(size_t)Whhp;
    for (int i = threadIdx.x; i < 8192; i += 128) {       // 8192 * 16B = 128 KB each
      unsigned loff0 = lbase + (unsigned)i * 16u;
      unsigned long long ga = g0 + (unsigned long long)i * 16ull;
      asm volatile("global_load_async_to_lds_b128 %0, %1, off"
                   :: "v"(loff0), "v"(ga) : "memory");
      unsigned loff1 = loff0 + 131072u;
      unsigned long long gb = g1 + (unsigned long long)i * 16ull;
      asm volatile("global_load_async_to_lds_b128 %0, %1, off"
                   :: "v"(loff1), "v"(gb) : "memory");
    }
    asm volatile("s_wait_asynccnt 0x0" ::: "memory");
  }
  const u16* WihL = wlds;
  const u16* WhhL = wlds + 65536;

  for (int i = threadIdx.x; i < 16 * Hq; i += 128) { hst[i] = 0; cst[i] = 0.f; }
  __syncthreads();

  for (int t = 0; t < Tq; ++t) {
    if (threadIdx.x < 16)                              // stage this step's row masks
      ems[threadIdx.x] = em[(row0 + threadIdx.x) * (long)Tq + t];
    const u16* xrow = Xg + ((row0 + lrow) * Tq + t) * (long)Hq;
    const u16* hrow = hst + lrow * Hq;
    // preload A fragments once per step (reused by all 8 gate N-tiles)
    Frag xa[4], ha[4];
    #pragma unroll
    for (int kt = 0; kt < 4; ++kt) {
      const v4u* ap = (const v4u*)(xrow + kt * 32 + half * 8);
      xa[kt].q[0] = ap[0]; xa[kt].q[1] = ap[2];
      const v4u* hp = (const v4u*)(hrow + kt * 32 + half * 8);
      ha[kt].q[0] = hp[0]; ha[kt].q[1] = hp[2];
    }
    for (int ntl = 0; ntl < 8; ++ntl) {
      int nt = wave * 8 + ntl;                         // N-tile over 512 gate cols
      // preload ALL 8 B fragments (4 ih + 4 hh) -> ds_loads pipeline ahead
      Frag bfr[8];
      #pragma unroll
      for (int kt = 0; kt < 4; ++kt) {
        const v4u* bp = (const v4u*)(WihL + (((size_t)(nt * 4 + kt) * 32 + lane) << 4));
        bfr[kt].q[0] = bp[0]; bfr[kt].q[1] = bp[1];
        const v4u* hp = (const v4u*)(WhhL + (((size_t)(nt * 4 + kt) * 32 + lane) << 4));
        bfr[4 + kt].q[0] = hp[0]; bfr[4 + kt].q[1] = hp[1];
      }
      v8f acc;
      float bias = bih[nt * 16 + lrow] + bhh[nt * 16 + lrow];
      #pragma unroll
      for (int i = 0; i < 8; ++i) acc[i] = bias;
      #pragma unroll
      for (int kt = 0; kt < 4; ++kt)                   // x @ W_ih^T
        acc = __builtin_amdgcn_wmma_f32_16x16x32_bf16(false, xa[kt].v, false, bfr[kt].v,
                                                      (short)0, acc, false, false);
      #pragma unroll
      for (int kt = 0; kt < 4; ++kt)                   // h @ W_hh^T
        acc = __builtin_amdgcn_wmma_f32_16x16x32_bf16(false, ha[kt].v, false, bfr[4 + kt].v,
                                                      (short)0, acc, false, false);
      // wave-uniform nonlinearity selection (scalar branch, no exec ping-pong)
      float* gout = &gbuf[wave][(half * 8) * Hq + ntl * 16 + lrow];
      if (wave == 2) {
        #pragma unroll
        for (int v = 0; v < 8; ++v) gout[v * Hq] = tanh_fast(acc[v]);
      } else {
        #pragma unroll
        for (int v = 0; v < 8; ++v) gout[v * Hq] = sig_fast(acc[v]);
      }
    }
    __syncthreads();
    #pragma unroll 4
    for (int e = 0; e < 16; ++e) {                     // masked c/h update
      int pos = e * 128 + threadIdx.x;
      int row = pos >> 7, col = pos & 127;
      float ig = gbuf[0][pos], fg = gbuf[1][pos];
      float gg = gbuf[2][pos], og = gbuf[3][pos];
      float c  = cst[pos];
      float cn = fg * c + ig * gg;
      float hn = og * tanh_fast(cn);
      long rt = (row0 + row) * (long)Tq + t;
      fe[rt * Hq + col] = f2bf(hn);                    // scan output: h_new (unmasked)
      if (ems[row]) { cst[pos] = cn; hst[pos] = f2bf(hn); }
    }
    __syncthreads();
  }
}

// logits = P @ Wo(128x4) + bo  (fp32 out)
__global__ void final_proj(const u16* __restrict__ P, const float* __restrict__ Wo,
                           const float* __restrict__ bo, float* __restrict__ out,
                           long rows) {
  long r = (long)blockIdx.x * blockDim.x + threadIdx.x;
  if (r >= rows) return;
  float a0 = bo[0], a1 = bo[1], a2 = bo[2], a3 = bo[3];
  const u16* prow = P + r * Hq;
  for (int c = 0; c < Hq; ++c) {
    float p = bf2f(prow[c]);
    a0 += p * Wo[c * 4 + 0]; a1 += p * Wo[c * 4 + 1];
    a2 += p * Wo[c * 4 + 2]; a3 += p * Wo[c * 4 + 3];
  }
  out[r * 4 + 0] = a0; out[r * 4 + 1] = a1;
  out[r * 4 + 2] = a2; out[r * 4 + 3] = a3;
}

template<int K>
static void launch_mlp(const u16* X, long rows, const u16* W1p, const float* b1,
                       const u16* W2p, const float* b2, u16* Y, hipStream_t s) {
  long tiles = rows >> 4;
  mlp2_wmma<K><<<(unsigned)((tiles + 3) / 4), 128, 0, s>>>(X, rows, W1p, b1, W2p, b2, Y);
}

// ---------------------------------------------------------------------------
extern "C" void kernel_launch(void* const* d_in, const int* in_sizes, int n_in,
                              void* d_out, int out_size, void* d_ws, size_t ws_size,
                              hipStream_t stream) {
  // Input order: trajectories, trajectory_mask, then params leaves in sorted
  // pytree order: edge1{b1,b2,w1,w2}, edge2{..}, lstm{b_hh,b_ih,w_hh,w_ih},
  // node1{..}, node2{..}, prior{..}, prior_out_b, prior_out_w.
  const float* traj     = (const float*)d_in[0];
  const u8*    mask     = (const u8*)   d_in[1];
  const float* e1_b1    = (const float*)d_in[2];
  const float* e1_b2    = (const float*)d_in[3];
  const float* e1_w1    = (const float*)d_in[4];
  const float* e1_w2    = (const float*)d_in[5];
  const float* e2_b1    = (const float*)d_in[6];
  const float* e2_b2    = (const float*)d_in[7];
  const float* e2_w1    = (const float*)d_in[8];
  const float* e2_w2    = (const float*)d_in[9];
  const float* l_bhh    = (const float*)d_in[10];
  const float* l_bih    = (const float*)d_in[11];
  const float* l_whh    = (const float*)d_in[12];
  const float* l_wih    = (const float*)d_in[13];
  const float* n1_b1    = (const float*)d_in[14];
  const float* n1_b2    = (const float*)d_in[15];
  const float* n1_w1    = (const float*)d_in[16];
  const float* n1_w2    = (const float*)d_in[17];
  const float* n2_b1    = (const float*)d_in[18];
  const float* n2_b2    = (const float*)d_in[19];
  const float* n2_w1    = (const float*)d_in[20];
  const float* n2_w2    = (const float*)d_in[21];
  const float* pr_b1    = (const float*)d_in[22];
  const float* pr_b2    = (const float*)d_in[23];
  const float* pr_w1    = (const float*)d_in[24];
  const float* pr_w2    = (const float*)d_in[25];
  const float* pr_ob    = (const float*)d_in[26];
  const float* pr_ow    = (const float*)d_in[27];
  float* out = (float*)d_out;

  const long rows_n = (long)Bq * Aq * Tq;   // 12800
  const long rows_e = (long)Bq * Eq * Tq;   // 396800

  char* ws = (char*)d_ws;
  size_t off = 0;
  auto alloc = [&](size_t bytes) -> void* {
    void* p = ws + off;
    off = (off + bytes + 255) & ~(size_t)255;
    return p;
  };

  // packed weights (bf16)
  u16* pw_n1_1 = (u16*)alloc((size_t)32  * 128 * 2);
  u16* pw_n1_2 = (u16*)alloc((size_t)128 * 128 * 2);
  u16* pw_e1_1 = (u16*)alloc((size_t)256 * 128 * 2);
  u16* pw_e1_2 = (u16*)alloc((size_t)128 * 128 * 2);
  u16* pw_n2_1 = (u16*)alloc((size_t)128 * 128 * 2);
  u16* pw_n2_2 = (u16*)alloc((size_t)128 * 128 * 2);
  u16* pw_e2_1 = (u16*)alloc((size_t)384 * 128 * 2);
  u16* pw_e2_2 = (u16*)alloc((size_t)128 * 128 * 2);
  u16* pw_pr_1 = (u16*)alloc((size_t)128 * 128 * 2);
  u16* pw_pr_2 = (u16*)alloc((size_t)128 * 128 * 2);
  u16* pw_ih   = (u16*)alloc((size_t)128 * 512 * 2);
  u16* pw_hh   = (u16*)alloc((size_t)128 * 512 * 2);

  // activations
  u16* X0   = (u16*)alloc((size_t)rows_n * 32  * 2);
  u16* n1b  = (u16*)alloc((size_t)rows_n * 128 * 2);
  u8*  em   = (u8*) alloc((size_t)rows_e);
  u16* regA = (u16*)alloc((size_t)rows_e * 384 * 2);  // Xe1 then Xe2
  u16* e1b  = (u16*)alloc((size_t)rows_e * 128 * 2);  // edge1 out; later reused for prior out
  u16* aggb = (u16*)alloc((size_t)rows_n * 128 * 2);
  u16* n2b  = (u16*)alloc((size_t)rows_n * 128 * 2);
  u16* e2b  = (u16*)alloc((size_t)rows_e * 128 * 2);  // edge2 out = LSTM x
  u16* feb  = (u16*)alloc((size_t)rows_e * 128 * 2);  // LSTM out
  u16* Pb   = e1b;                                    // prior MLP out (reuse)

  auto pack = [&](const float* W, u16* Wp, int K, int N, long sk, long sn, int Kpad) {
    int total = (N / 16) * (Kpad / 32) * 512;
    pack_w<<<(total + 255) / 256, 256, 0, stream>>>(W, Wp, K, N, sk, sn, Kpad);
  };
  pack(n1_w1, pw_n1_1,   8, 128, 128, 1,  32);
  pack(n1_w2, pw_n1_2, 128, 128, 128, 1, 128);
  pack(e1_w1, pw_e1_1, 256, 128, 128, 1, 256);
  pack(e1_w2, pw_e1_2, 128, 128, 128, 1, 128);
  pack(n2_w1, pw_n2_1, 128, 128, 128, 1, 128);
  pack(n2_w2, pw_n2_2, 128, 128, 128, 1, 128);
  pack(e2_w1, pw_e2_1, 384, 128, 128, 1, 384);
  pack(e2_w2, pw_e2_2, 128, 128, 128, 1, 128);
  pack(pr_w1, pw_pr_1, 128, 128, 128, 1, 128);
  pack(pr_w2, pw_pr_2, 128, 128, 128, 1, 128);
  pack(l_wih, pw_ih,   128, 512, 1, 128, 128);   // W_ih^T: (128 x 512)
  pack(l_whh, pw_hh,   128, 512, 1, 128, 128);   // W_hh^T

  build_x0<<<(unsigned)((rows_n * 32 + 255) / 256), 256, 0, stream>>>(traj, X0, rows_n);
  make_em <<<(unsigned)((rows_e + 255) / 256), 256, 0, stream>>>(mask, em);

  // node1
  launch_mlp<32>(X0, rows_n, pw_n1_1, n1_b1, pw_n1_2, n1_b2, n1b, stream);
  // edge1
  concat_edge1<<<(unsigned)((rows_e * 256 + 255) / 256), 256, 0, stream>>>(n1b, regA);
  launch_mlp<256>(regA, rows_e, pw_e1_1, e1_b1, pw_e1_2, e1_b2, e1b, stream);
  // aggregate -> node2
  agg_edges<<<(unsigned)((rows_n * 128 + 255) / 256), 256, 0, stream>>>(e1b, em, aggb);
  launch_mlp<128>(aggb, rows_n, pw_n2_1, n2_b1, pw_n2_2, n2_b2, n2b, stream);
  // edge2
  concat_edge2<<<(unsigned)((rows_e * 384 + 255) / 256), 256, 0, stream>>>(n2b, e1b, regA);
  launch_mlp<384>(regA, rows_e, pw_e2_1, e2_b1, pw_e2_2, e2_b2, e2b, stream);
  // LSTM over T (persistent h/c in LDS; weights staged to 256 KB dynamic LDS)
  lstm_wmma<<<BEq / 16, 128, 262144, stream>>>(e2b, em, pw_ih, pw_hh, l_bih, l_bhh, feb);
  // prior MLP + output projection
  launch_mlp<128>(feb, rows_e, pw_pr_1, pr_b1, pw_pr_2, pr_b2, Pb, stream);
  final_proj<<<(unsigned)((rows_e + 255) / 256), 256, 0, stream>>>(Pb, pr_ow, pr_ob, out, rows_e);
}
